// StateSpace2D_58471684768286
// MI455X (gfx1250) — compile-verified, matrix operand order checked by reading, not run on hardware
//
#include <hip/hip_runtime.h>

#define DECAY 0.9f
#define BN_EPS 1e-5f
#define CC 256
#define HH 128
#define WW 128
#define HW 16384          // 128*128
#define BB 8
#define NPIX_PER_CH 131072  // B*H*W

typedef __attribute__((ext_vector_type(16))) __bf16 v16bf;
typedef __attribute__((ext_vector_type(8)))  float  v8f;

union BFrag {
    uint4 u[2];
    v16bf v;
};

__device__ __forceinline__ unsigned short f2bf(float f) {
    unsigned u = __builtin_bit_cast(unsigned, f);
    u += 0x7FFFu + ((u >> 16) & 1u);           // round-to-nearest-even
    return (unsigned short)(u >> 16);
}

// ---------------------------------------------------------------------------
// K0: convert the three [256,256] fp32 weights to bf16; zero BN stats buffer.
// ---------------------------------------------------------------------------
__global__ __launch_bounds__(256) void cvt_weights(
    const float* __restrict__ w0, const float* __restrict__ w1,
    const float* __restrict__ w2, unsigned short* __restrict__ wbf,
    float* __restrict__ stats) {
    int i = blockIdx.x * 256 + threadIdx.x;    // 0..65535
    wbf[i]           = f2bf(w0[i]);
    wbf[65536 + i]   = f2bf(w1[i]);
    wbf[131072 + i]  = f2bf(w2[i]);
    if (i < 512) stats[i] = 0.0f;
}

// ---------------------------------------------------------------------------
// K1/K5: out[o,p] = sum_c W[o,c] * in[c,p], per batch. bf16 WMMA, fp32 accum.
// Block = 8 waves; block tile = 256 O x 64 pixels; K step = 32.
// ---------------------------------------------------------------------------
__global__ __launch_bounds__(256) void gemm256(
    const float* __restrict__ in, const unsigned short* __restrict__ wbf,
    float* __restrict__ out) {
    __shared__ unsigned short ldsX[64 * 40];   // [pixel][K] bf16, stride 40 halves (80B)

    int blk   = blockIdx.x;
    int batch = blk >> 8;                      // 256 pixel tiles per batch
    int pBase = (blk & 255) << 6;              // *64 pixels
    size_t batchBase = (size_t)batch * CC * HW;

    int tid  = threadIdx.x;
    int wave = tid >> 5;
    int lane = tid & 31;
    int g    = lane >> 4;
    int lm   = lane & 15;
    int o0   = wave * 32;

    v8f z = {0.f,0.f,0.f,0.f,0.f,0.f,0.f,0.f};
    v8f acc[2][4];
#pragma unroll
    for (int i = 0; i < 2; i++)
#pragma unroll
        for (int j = 0; j < 4; j++) acc[i][j] = z;

    int kRow  = tid >> 3;                      // 0..31  (K row within step)
    int chunk = tid & 7;                       // 8 pixels each -> coalesced 256B/row

    for (int kb = 0; kb < CC; kb += 32) {
        const float* src = in + batchBase + (size_t)(kb + kRow) * HW + pBase + chunk * 8;
        float4 f0 = *(const float4*)src;
        float4 f1 = *(const float4*)(src + 4);
        int p0 = chunk * 8;
        ldsX[(p0 + 0) * 40 + kRow] = f2bf(f0.x);
        ldsX[(p0 + 1) * 40 + kRow] = f2bf(f0.y);
        ldsX[(p0 + 2) * 40 + kRow] = f2bf(f0.z);
        ldsX[(p0 + 3) * 40 + kRow] = f2bf(f0.w);
        ldsX[(p0 + 4) * 40 + kRow] = f2bf(f1.x);
        ldsX[(p0 + 5) * 40 + kRow] = f2bf(f1.y);
        ldsX[(p0 + 6) * 40 + kRow] = f2bf(f1.z);
        ldsX[(p0 + 7) * 40 + kRow] = f2bf(f1.w);
        __syncthreads();

        // A fragments: W rows, two 16B runs per lane (K = kb+8g..+7, kb+16+8g..+23)
        BFrag a0, a1, bf;
        const uint4* w0p = (const uint4*)(wbf + (size_t)(o0 + lm) * CC + kb + 8 * g);
        a0.u[0] = w0p[0]; a0.u[1] = w0p[2];
        const uint4* w1p = (const uint4*)(wbf + (size_t)(o0 + 16 + lm) * CC + kb + 8 * g);
        a1.u[0] = w1p[0]; a1.u[1] = w1p[2];

#pragma unroll
        for (int pt = 0; pt < 4; pt++) {
            const uint4* bp = (const uint4*)(&ldsX[(pt * 16 + lm) * 40 + 8 * g]);
            bf.u[0] = bp[0]; bf.u[1] = bp[2];
            acc[0][pt] = __builtin_amdgcn_wmma_f32_16x16x32_bf16(
                false, a0.v, false, bf.v, (short)0, acc[0][pt], false, false);
            acc[1][pt] = __builtin_amdgcn_wmma_f32_16x16x32_bf16(
                false, a1.v, false, bf.v, (short)0, acc[1][pt], false, false);
        }
        __syncthreads();
    }

#pragma unroll
    for (int ot = 0; ot < 2; ot++)
#pragma unroll
        for (int pt = 0; pt < 4; pt++)
#pragma unroll
            for (int r = 0; r < 8; r++) {
                int o = o0 + ot * 16 + 8 * g + r;
                int p = pBase + pt * 16 + lm;
                out[batchBase + (size_t)o * HW + p] = acc[ot][pt][r];
            }
}

// ---------------------------------------------------------------------------
// K2: EMA scan along W. One thread per (b,c,h) row.
// ---------------------------------------------------------------------------
__global__ __launch_bounds__(256) void scan_w_k(
    const float* __restrict__ xp, float* __restrict__ outp) {
    size_t row  = (size_t)blockIdx.x * 256 + threadIdx.x;   // < 262144
    size_t base = row * WW;
    float y = 0.0f;
    for (int w = 0; w < WW; w++) {
        y = DECAY * y + xp[base + w];
        outp[base + w] = y;
    }
}

// ---------------------------------------------------------------------------
// K3: EMA scan along H, accumulated into the scan buffer (coalesced columns).
// ---------------------------------------------------------------------------
__global__ __launch_bounds__(256) void scan_h_k(
    const float* __restrict__ xp, float* __restrict__ outp) {
    size_t col   = (size_t)blockIdx.x * 256 + threadIdx.x;  // < 262144
    size_t plane = col >> 7;
    int    w     = (int)(col & 127);
    size_t base  = plane * HW + w;
    float y = 0.0f;
    for (int h = 0; h < HH; h++) {
        size_t idx = base + (size_t)h * WW;
        y = DECAY * y + xp[idx];
        outp[idx] += y;
    }
}

// ---------------------------------------------------------------------------
// K4: gate GEMM (WMMA) + sigmoid + depthwise 3x3 + blend. In-place over sf.
// ---------------------------------------------------------------------------
__global__ __launch_bounds__(256) void gate_fuse(
    const float* __restrict__ xp, const unsigned short* __restrict__ wg,
    const float* __restrict__ gate_b, const float* __restrict__ dww,
    float* __restrict__ sf) {
    __shared__ unsigned short ldsX[64 * 40];

    int blk   = blockIdx.x;
    int batch = blk >> 8;
    int pBase = (blk & 255) << 6;
    size_t batchBase = (size_t)batch * CC * HW;

    int tid  = threadIdx.x;
    int wave = tid >> 5;
    int lane = tid & 31;
    int g    = lane >> 4;
    int lm   = lane & 15;
    int o0   = wave * 32;

    v8f z = {0.f,0.f,0.f,0.f,0.f,0.f,0.f,0.f};
    v8f acc[2][4];
#pragma unroll
    for (int i = 0; i < 2; i++)
#pragma unroll
        for (int j = 0; j < 4; j++) acc[i][j] = z;

    int kRow  = tid >> 3;
    int chunk = tid & 7;

    for (int kb = 0; kb < CC; kb += 32) {
        const float* src = xp + batchBase + (size_t)(kb + kRow) * HW + pBase + chunk * 8;
        float4 f0 = *(const float4*)src;
        float4 f1 = *(const float4*)(src + 4);
        int p0 = chunk * 8;
        ldsX[(p0 + 0) * 40 + kRow] = f2bf(f0.x);
        ldsX[(p0 + 1) * 40 + kRow] = f2bf(f0.y);
        ldsX[(p0 + 2) * 40 + kRow] = f2bf(f0.z);
        ldsX[(p0 + 3) * 40 + kRow] = f2bf(f0.w);
        ldsX[(p0 + 4) * 40 + kRow] = f2bf(f1.x);
        ldsX[(p0 + 5) * 40 + kRow] = f2bf(f1.y);
        ldsX[(p0 + 6) * 40 + kRow] = f2bf(f1.z);
        ldsX[(p0 + 7) * 40 + kRow] = f2bf(f1.w);
        __syncthreads();

        BFrag a0, a1, bf;
        const uint4* w0p = (const uint4*)(wg + (size_t)(o0 + lm) * CC + kb + 8 * g);
        a0.u[0] = w0p[0]; a0.u[1] = w0p[2];
        const uint4* w1p = (const uint4*)(wg + (size_t)(o0 + 16 + lm) * CC + kb + 8 * g);
        a1.u[0] = w1p[0]; a1.u[1] = w1p[2];

#pragma unroll
        for (int pt = 0; pt < 4; pt++) {
            const uint4* bp = (const uint4*)(&ldsX[(pt * 16 + lm) * 40 + 8 * g]);
            bf.u[0] = bp[0]; bf.u[1] = bp[2];
            acc[0][pt] = __builtin_amdgcn_wmma_f32_16x16x32_bf16(
                false, a0.v, false, bf.v, (short)0, acc[0][pt], false, false);
            acc[1][pt] = __builtin_amdgcn_wmma_f32_16x16x32_bf16(
                false, a1.v, false, bf.v, (short)0, acc[1][pt], false, false);
        }
        __syncthreads();
    }

    // Epilogue: gate + depthwise 3x3 + blend, in-place into sf.
#pragma unroll
    for (int ot = 0; ot < 2; ot++) {
#pragma unroll
        for (int r = 0; r < 8; r++) {
            int o = o0 + ot * 16 + 8 * g + r;
            float bias = gate_b[o];
            const float* dwr = dww + o * 9;
            float d00 = dwr[0], d01 = dwr[1], d02 = dwr[2];
            float d10 = dwr[3], d11 = dwr[4], d12 = dwr[5];
            float d20 = dwr[6], d21 = dwr[7], d22 = dwr[8];
            size_t cbase = batchBase + (size_t)o * HW;
#pragma unroll
            for (int pt = 0; pt < 4; pt++) {
                int p = pBase + pt * 16 + lm;
                int h = p >> 7, w = p & 127;
                float gpre = acc[ot][pt][r] + bias;
                float gate = 1.0f / (1.0f + __expf(-gpre));
                float xv = xp[cbase + p];
                float loc = d11 * xv;
                const float* rowm = xp + cbase + p - WW;
                const float* row0 = xp + cbase + p;
                const float* rowp = xp + cbase + p + WW;
                bool wl = (w > 0), wh = (w < WW - 1);
                if (h > 0) {
                    loc += d01 * rowm[0];
                    if (wl) loc += d00 * rowm[-1];
                    if (wh) loc += d02 * rowm[1];
                }
                if (wl) loc += d10 * row0[-1];
                if (wh) loc += d12 * row0[1];
                if (h < HH - 1) {
                    loc += d21 * rowp[0];
                    if (wl) loc += d20 * rowp[-1];
                    if (wh) loc += d22 * rowp[1];
                }
                float ss = sf[cbase + p];            // scan_w + scan_h
                float fused = gate * (loc + ss) * (1.0f / 3.0f) + (1.0f - gate) * xv;
                sf[cbase + p] = fused;
            }
        }
    }
}

// ---------------------------------------------------------------------------
// K6: per-channel sum / sumsq over (B,H,W). Block per (channel,batch).
// ---------------------------------------------------------------------------
__global__ __launch_bounds__(256) void bn_stats(
    const float* __restrict__ y, float* __restrict__ stats) {
    int ch = blockIdx.x >> 3;
    int b  = blockIdx.x & 7;
    size_t base = ((size_t)b * CC + ch) * HW;
    int tid = threadIdx.x;
    float s = 0.0f, q = 0.0f;
    for (int i = tid; i < HW; i += 256) {
        float v = y[base + i];
        s += v;
        q += v * v;
    }
    __shared__ float sh[512];
    sh[tid] = s;
    sh[256 + tid] = q;
    __syncthreads();
    for (int st = 128; st > 0; st >>= 1) {
        if (tid < st) {
            sh[tid] += sh[tid + st];
            sh[256 + tid] += sh[256 + tid + st];
        }
        __syncthreads();
    }
    if (tid == 0) {
        atomicAdd(&stats[ch], sh[0]);
        atomicAdd(&stats[256 + ch], sh[256]);
    }
}

// ---------------------------------------------------------------------------
// K7: out = x + gamma*(y-mean)*rsqrt(var+eps) + beta   (float4 vectorized)
// ---------------------------------------------------------------------------
__global__ __launch_bounds__(256) void finalize_k(
    const float4* __restrict__ x4, const float4* __restrict__ y4,
    const float* __restrict__ stats, const float* __restrict__ gamma,
    const float* __restrict__ beta, float4* __restrict__ out4) {
    size_t i = (size_t)blockIdx.x * 256 + threadIdx.x;   // < 8388608
    int ch = (int)((i >> 12) & 255);                     // (i*4 / 16384) % 256
    float mean = stats[ch] * (1.0f / (float)NPIX_PER_CH);
    float var  = stats[256 + ch] * (1.0f / (float)NPIX_PER_CH) - mean * mean;
    float inv  = rsqrtf(var + BN_EPS);
    float a = gamma[ch] * inv;
    float b = beta[ch] - mean * a;
    float4 xv = x4[i], yv = y4[i], o;
    o.x = xv.x + yv.x * a + b;
    o.y = xv.y + yv.y * a + b;
    o.z = xv.z + yv.z * a + b;
    o.w = xv.w + yv.w * a + b;
    out4[i] = o;
}

extern "C" void kernel_launch(void* const* d_in, const int* in_sizes, int n_in,
                              void* d_out, int out_size, void* d_ws, size_t ws_size,
                              hipStream_t stream) {
    const float* x      = (const float*)d_in[0];
    const float* in_w   = (const float*)d_in[1];
    const float* gate_w = (const float*)d_in[2];
    const float* gate_b = (const float*)d_in[3];
    const float* dw_w   = (const float*)d_in[4];
    const float* out_w  = (const float*)d_in[5];
    const float* gamma  = (const float*)d_in[6];
    const float* beta   = (const float*)d_in[7];
    float* out = (float*)d_out;

    char* ws = (char*)d_ws;
    const size_t TEN_BYTES = (size_t)BB * CC * HW * sizeof(float);   // 134217728
    float*          ws0   = (float*)ws;                         // x_proj, then y
    float*          ws1   = (float*)(ws + TEN_BYTES);           // scan sum, then fused
    unsigned short* wbf   = (unsigned short*)(ws + 2 * TEN_BYTES);   // 3*65536 bf16
    float*          stats = (float*)(ws + 2 * TEN_BYTES + 393216);   // 512 floats

    cvt_weights<<<256, 256, 0, stream>>>(in_w, gate_w, out_w, wbf, stats);
    gemm256   <<<2048, 256, 0, stream>>>(x, wbf, ws0);                 // x_proj
    scan_w_k  <<<1024, 256, 0, stream>>>(ws0, ws1);                    // scan_w
    scan_h_k  <<<1024, 256, 0, stream>>>(ws0, ws1);                    // += scan_h
    gate_fuse <<<2048, 256, 0, stream>>>(ws0, wbf + 65536, gate_b, dw_w, ws1);
    gemm256   <<<2048, 256, 0, stream>>>(ws1, wbf + 131072, ws0);      // y = out_w·fused
    bn_stats  <<<2048, 256, 0, stream>>>(ws0, stats);
    finalize_k<<<32768, 256, 0, stream>>>((const float4*)x, (const float4*)ws0,
                                          stats, gamma, beta, (float4*)out);
}
